// C2Q_15032385536516
// MI455X (gfx1250) — matrix operand order repaired and checked by reading, not compiled, not run on hardware
//
#include <hip/hip_runtime.h>
#include <hip/hip_bf16.h>

typedef __attribute__((ext_vector_type(16))) __bf16    v16bf;
typedef __attribute__((ext_vector_type(8)))  float     v8f;
typedef __attribute__((ext_vector_type(4)))  float     v4f;
typedef __attribute__((ext_vector_type(4)))  unsigned  v4u;

#define TILE_T  64
#define JDIM    128
#define DDIM    512
#define TDIM    2048
#define BDIM    128
#define ASTRIDE 136   // bf16 elems per a_lds row: 128 + 8 pad (272B, 16B-aligned rows)

union AFrag { v4u q[2]; v16bf v; };

__global__ __launch_bounds__(256)
void C2Q_softmax_attend_kernel(const float* __restrict__ U,
                               const float* __restrict__ S,
                               float* __restrict__ Out) {
    __shared__ __align__(16) __bf16 a_lds[TILE_T * ASTRIDE];

    const int tid = threadIdx.x;
    const int tBlocks = TDIM / TILE_T;          // 32
    const int b  = blockIdx.x / tBlocks;
    const int t0 = (blockIdx.x % tBlocks) * TILE_T;

    // ---------------- Phase 1: row softmax over J, result -> bf16 in LDS ----
    {
        const int r = tid >> 2;                 // 0..63 : T-row within tile
        const int q = tid & 3;                  // 0..3  : 32-col slice of J
        const float* srow = S + ((size_t)(b * TDIM + t0 + r)) * JDIM + q * 32;
        float v[32];
        #pragma unroll
        for (int i = 0; i < 8; ++i) {
            v4f f = __builtin_nontemporal_load(((const v4f*)srow) + i);
            v[4*i+0] = f.x; v[4*i+1] = f.y; v[4*i+2] = f.z; v[4*i+3] = f.w;
        }
        float m = v[0];
        #pragma unroll
        for (int i = 1; i < 32; ++i) m = fmaxf(m, v[i]);
        m = fmaxf(m, __shfl_xor(m, 1, 32));     // quad lives inside one wave32
        m = fmaxf(m, __shfl_xor(m, 2, 32));
        float sum = 0.f;
        #pragma unroll
        for (int i = 0; i < 32; ++i) { v[i] = __expf(v[i] - m); sum += v[i]; }
        sum += __shfl_xor(sum, 1, 32);
        sum += __shfl_xor(sum, 2, 32);
        const float inv = __builtin_amdgcn_rcpf(sum);
        __bf16* arow = &a_lds[r * ASTRIDE + q * 32];
        #pragma unroll
        for (int i = 0; i < 32; ++i) arow[i] = (__bf16)(v[i] * inv);
    }
    __syncthreads();

    // ---------------- Phase 2: out = a @ U[b]  via v_wmma_f32_16x16x32_bf16 --
    const int wave = tid >> 5;                  // 0..7 -> owns 64 d-columns
    const int lane = tid & 31;
    const int g    = lane >> 4;                 // half-wave group
    const int ln   = lane & 15;

    // Preload all A fragments afr[mt][kt] (16-bit A 16x32 layout):
    // elem i, lane group g: K = (i>>3)*16 + g*8 + (i&7); M = lane&15.
    AFrag afr[4][4];
    #pragma unroll
    for (int mt = 0; mt < 4; ++mt) {
        const __bf16* mrow = &a_lds[(mt * 16 + ln) * ASTRIDE + g * 8];
        #pragma unroll
        for (int kt = 0; kt < 4; ++kt) {
            afr[mt][kt].q[0] = *(const v4u*)(mrow + kt * 32);       // K  0.. 7 (+g*8)
            afr[mt][kt].q[1] = *(const v4u*)(mrow + kt * 32 + 16);  // K 16..23 (+g*8)
        }
    }

    const float* Ub = U   + (size_t)b * JDIM * DDIM;
    float*       Ob = Out + (size_t)b * TDIM * DDIM;
    const int n_base_w = wave * 64;

    #pragma unroll
    for (int nt = 0; nt < 4; ++nt) {
        const int n = n_base_w + nt * 16 + ln;  // N = lane&15
        // B fragments (32x16, K x N): elem i, lane group g: K = (i>>3)*16 + g*8 + (i&7)
        v16bf bfr[4];
        #pragma unroll
        for (int kt = 0; kt < 4; ++kt) {
            const int kb = kt * 32 + g * 8;
            #pragma unroll
            for (int i = 0; i < 16; ++i)
                bfr[kt][i] = (__bf16)Ub[(size_t)(kb + (i & 7) + ((i >> 3) << 4)) * DDIM + n];
        }
        #pragma unroll
        for (int mt = 0; mt < 4; ++mt) {
            v8f acc = {};
            #pragma unroll
            for (int kt = 0; kt < 4; ++kt) {
                acc = __builtin_amdgcn_wmma_f32_16x16x32_bf16(
                    false, afr[mt][kt].v, false, bfr[kt],
                    (short)0, acc, false, false);
            }
            // C/D layout: VGPR r -> M = r + g*8, N = lane&15
            float* orow = Ob + (size_t)(t0 + mt * 16 + g * 8) * DDIM + n;
            #pragma unroll
            for (int r = 0; r < 8; ++r)
                __builtin_nontemporal_store(acc[r], orow + (size_t)r * DDIM);
        }
    }
}

extern "C" void kernel_launch(void* const* d_in, const int* in_sizes, int n_in,
                              void* d_out, int out_size, void* d_ws, size_t ws_size,
                              hipStream_t stream) {
    const float* U = (const float*)d_in[0];   // [B, J, d] f32
    const float* S = (const float*)d_in[1];   // [B, T, J] f32
    float* Out = (float*)d_out;               // [B, T, d] f32
    const int grid = BDIM * (TDIM / TILE_T);  // 128 * 32 = 4096
    C2Q_softmax_attend_kernel<<<grid, 256, 0, stream>>>(U, S, Out);
}
